// OnlineLSTM_24730421690617
// MI455X (gfx1250) — matrix operand order repaired and checked by reading, not compiled, not run on hardware
//
#include <hip/hip_runtime.h>

typedef __attribute__((ext_vector_type(16))) _Float16 v16h;
typedef __attribute__((ext_vector_type(8)))  _Float16 v8h;
typedef __attribute__((ext_vector_type(8)))  float    v8f;

#define LSTM_T   2048
#define LSTM_H   50
#define NTILE_P  16      // padded N-tiles: 16 x 16 = 256 cols (200 used) -> branch-free
#define GP       256
#define KP       64      // K padded: 50 h-cols + [1.0] + [x_t] + zeros
#define WAVES    4       // one 16-row batch tile per WG; 4 N-tiles per wave

// ---- CDNA5 hardware tanh (TRANS32) with graceful fallback ----
#if __has_builtin(__builtin_amdgcn_tanhf)
__device__ __forceinline__ float tanh_f(float x) { return __builtin_amdgcn_tanhf(x); }
#elif __has_builtin(__builtin_amdgcn_tanh_f32)
__device__ __forceinline__ float tanh_f(float x) { return __builtin_amdgcn_tanh_f32(x); }
#else
__device__ __forceinline__ float tanh_f(float x) {
    float e = __expf(2.0f * x);
    return 1.0f - 2.0f * __builtin_amdgcn_rcpf(e + 1.0f);
}
#endif
// sigmoid via tanh: 3 VALU ops (mul, v_tanh, fma)
__device__ __forceinline__ float sigmoid_f(float x) {
    return __builtin_fmaf(0.5f, tanh_f(0.5f * x), 0.5f);
}

__global__ __launch_bounds__(WAVES * 32)
void lstm_wmma_gfx1250_kernel(const float* __restrict__ x,
                              const float* __restrict__ W_ih,
                              const float* __restrict__ W_hh,
                              const float* __restrict__ b_ih,
                              const float* __restrict__ b_hh,
                              const float* __restrict__ W_lin,
                              const float* __restrict__ b_lin,
                              float* __restrict__ out)
{
    __shared__ __align__(16) float    ldsZ[16][GP];   // 16 KB : z scratch (whole WG)
    __shared__ __align__(16) _Float16 ldsH[16][KP];   //  2 KB : h as A-matrix rows
    __shared__ __align__(16) float    ldsX[16][32];   //  2 KB : staged x
    __shared__ float ldsWlin[64];

    const int tid  = threadIdx.x;
    const int w    = __builtin_amdgcn_readfirstlane(tid >> 5);  // scalar wave id 0..3
    const int l    = tid & 31;
    const int half = l >> 4;          // 0 or 1
    const int m    = l & 15;
    const int rowbase = blockIdx.x * 16;

    // ---------------- one-time setup ----------------
    // B fragments of [W_hh^T ; bias ; W_ih ; 0] in REGISTERS for the whole kernel.
    // Wave w owns N-tiles {w, w+4, w+8, w+12}; tiles 13..15 are zero padding.
    // B k-order used here, same as A (32x16 KxN): lane l, elem j -> n = nt*16+(l&15), k = kf*32+j+16*(l>>4)
    v16h bf[4][2];
#pragma unroll
    for (int i = 0; i < 4; ++i) {
        const int n = (w + 4 * i) * 16 + m;
#pragma unroll
        for (int kf = 0; kf < 2; ++kf) {
#pragma unroll
            for (int j = 0; j < 16; ++j) {
                float val = 0.0f;
                if (n < 4 * LSTM_H) {
                    int k = kf * 32 + j + 16 * half;
                    if (k < LSTM_H)           val = W_hh[n * LSTM_H + k];
                    else if (k == LSTM_H)     val = b_ih[n] + b_hh[n]; // A col 50 == 1.0
                    else if (k == LSTM_H + 1) val = W_ih[n];           // A col 51 == x_t
                }
                bf[i][kf][j] = (_Float16)val;
            }
        }
    }
    // h0 = 0 ; A col 50 = 1.0 ; cols 51..63 = 0
    for (int idx = tid; idx < 16 * KP; idx += WAVES * 32) {
        int k = idx & (KP - 1);
        int r = idx >> 6;
        ldsH[r][k] = (k == LSTM_H) ? (_Float16)1.0f : (_Float16)0.0f;
    }
    if (tid < 64) ldsWlin[tid] = (tid < LSTM_H) ? W_lin[tid] : 0.0f;
    __syncthreads();

    // elementwise partition: row = tid&15, col = q + 8*jj with q = (tid>>4)&7.
    // jj = 0..5 -> col <= 47 (provably in range, no guards); tail col 48/49 for q<2.
    const int erow = tid & 15;
    const int q    = (tid >> 4) & 7;
    float cst[7];
#pragma unroll
    for (int j = 0; j < 7; ++j) cst[j] = 0.0f;

    const float* xrow = x + (size_t)rowbase * LSTM_T;

    for (int t = 0; t < LSTM_T; ++t) {
        // ---- stage x for 32 timesteps: 128 threads x 1 float4 = 16 rows x 32 ----
        if ((t & 31) == 0) {
            int r  = (tid >> 3) & 15;     // row 0..15
            int ch = tid & 7;             // float4 chunk 0..7
            float4 v = *(const float4*)(xrow + (size_t)r * LSTM_T + t + ch * 4);
            *(float4*)&ldsX[r][ch * 4] = v;
            if (t + 32 < LSTM_T)
                __builtin_prefetch(xrow + (size_t)erow * LSTM_T + t + 32, 0, 0);
            __syncthreads();
        }
        // x_t into A column 51
        if (tid < 16) ldsH[tid][LSTM_H + 1] = (_Float16)ldsX[tid][t & 31];
        __syncthreads();   // h (prev step) + x visible to all waves

        // ---- A fragments from h (16-bit A layout: lane m, K split per half) ----
        const _Float16* hb = &ldsH[m][0];
        v8h a0lo = *(const v8h*)(hb + 16 * half);
        v8h a0hi = *(const v8h*)(hb + 8 + 16 * half);
        v8h a1lo = *(const v8h*)(hb + 32 + 16 * half);
        v8h a1hi = *(const v8h*)(hb + 40 + 16 * half);
        v16h a0 = __builtin_shufflevector(a0lo, a0hi, 0,1,2,3,4,5,6,7,8,9,10,11,12,13,14,15);
        v16h a1 = __builtin_shufflevector(a1lo, a1hi, 0,1,2,3,4,5,6,7,8,9,10,11,12,13,14,15);

        // ---- z = [h 1 x] @ [W_hh^T ; bias ; W_ih] : 4 N-tiles, branch-free ----
#pragma unroll
        for (int i = 0; i < 4; ++i) {
            const int ncol = (w + 4 * i) * 16 + m;   // scalar base + lane offset
            v8f acc = {};
            acc = __builtin_amdgcn_wmma_f32_16x16x32_f16(false, a0, false, bf[i][0],
                                                         (short)0, acc, false, false);
            acc = __builtin_amdgcn_wmma_f32_16x16x32_f16(false, a1, false, bf[i][1],
                                                         (short)0, acc, false, false);
            // C layout: VGPR r, lanes 0-15 -> M=r ; lanes 16-31 -> M=8+r ; N=lane%16
#pragma unroll
            for (int r = 0; r < 8; ++r)
                ldsZ[r + 8 * half][ncol] = acc[r];
        }
        __syncthreads();   // z visible

        // ---- gates + state update: 768 cells unconditional + 32-cell tail ----
#pragma unroll
        for (int jj = 0; jj < 6; ++jj) {
            const int col = q + 8 * jj;               // <= 47 : provably valid
            float zi = ldsZ[erow][col];
            float zf = ldsZ[erow][LSTM_H + col];
            float zg = ldsZ[erow][2 * LSTM_H + col];
            float zo = ldsZ[erow][3 * LSTM_H + col];
            float ig = sigmoid_f(zi);
            float fg = sigmoid_f(zf);
            float gg = tanh_f(zg);
            float og = sigmoid_f(zo);
            float c  = fg * cst[jj] + ig * gg;
            cst[jj]  = c;
            ldsH[erow][col] = (_Float16)(og * tanh_f(c));
        }
        if (q < 2) {                                  // columns 48, 49
            const int col = 48 + q;
            float zi = ldsZ[erow][col];
            float zf = ldsZ[erow][LSTM_H + col];
            float zg = ldsZ[erow][2 * LSTM_H + col];
            float zo = ldsZ[erow][3 * LSTM_H + col];
            float ig = sigmoid_f(zi);
            float fg = sigmoid_f(zf);
            float gg = tanh_f(zg);
            float og = sigmoid_f(zo);
            float c  = fg * cst[6] + ig * gg;
            cst[6]   = c;
            ldsH[erow][col] = (_Float16)(og * tanh_f(c));
        }
        __syncthreads();   // h visible for next step's A loads
    }

    // ---------------- output: out[b] = h_last . W_lin + b_lin ----------------
    if (tid < 16) {
        float p = 0.0f;
#pragma unroll
        for (int j = 0; j < LSTM_H; ++j)
            p += (float)ldsH[tid][j] * ldsWlin[j];
        out[rowbase + tid] = p + b_lin[0];
    }
}

extern "C" void kernel_launch(void* const* d_in, const int* in_sizes, int n_in,
                              void* d_out, int out_size, void* d_ws, size_t ws_size,
                              hipStream_t stream) {
    const float* x     = (const float*)d_in[0];
    const float* W_ih  = (const float*)d_in[1];
    const float* W_hh  = (const float*)d_in[2];
    const float* b_ih  = (const float*)d_in[3];
    const float* b_hh  = (const float*)d_in[4];
    const float* W_lin = (const float*)d_in[5];
    const float* b_lin = (const float*)d_in[6];
    float* out = (float*)d_out;

    const int B    = in_sizes[0] / LSTM_T;     // I == 1
    const int grid = B / 16;                   // one 16-row batch tile per WG
    lstm_wmma_gfx1250_kernel<<<grid, WAVES * 32, 0, stream>>>(
        x, W_ih, W_hh, b_ih, b_hh, W_lin, b_lin, out);
}
